// MultiHeadAttention_19507741458806
// MI455X (gfx1250) — matrix operand order repaired and checked by reading, not compiled, not run on hardware
//
#include <hip/hip_runtime.h>
#include <math.h>

// ---------------------------------------------------------------------------
// Types for CDNA5 WMMA (wave32, v_wmma_f32_16x16x32_bf16)
// ---------------------------------------------------------------------------
typedef __attribute__((ext_vector_type(16))) __bf16 v16bf;
typedef __attribute__((ext_vector_type(8)))  float  v8f;
typedef unsigned short u16;
typedef __attribute__((ext_vector_type(8)))  u16    us8;
typedef __attribute__((ext_vector_type(4)))  unsigned v4u;
typedef __attribute__((ext_vector_type(4)))  int      v4i;
typedef __attribute__((ext_vector_type(8)))  int      v8i;

#define LDS_AS    __attribute__((address_space(3)))
#define GLOBAL_AS __attribute__((address_space(1)))

// Feature probes (all optional; fall back to plain VMEM/DS paths)
#if __has_builtin(__builtin_amdgcn_tensor_load_to_lds) && \
    __has_builtin(__builtin_amdgcn_s_wait_tensorcnt)
#define HAVE_TDM 1
#endif
#if __has_builtin(__builtin_amdgcn_global_load_async_to_lds_b128) && \
    __has_builtin(__builtin_amdgcn_s_wait_asynccnt)
#define HAVE_ASYNC 1
#endif

union FragBF {          // one WMMA A/B operand: 16 bf16 values per lane
    us8   h[2];         // two 16-byte halves (ds_load_b128 friendly)
    v16bf v;
};

__device__ __forceinline__ v8f wmma_bf16(v16bf a, v16bf b, v8f c) {
    // (neg_a, A, neg_b, B, c_mod, C, reuse_a, reuse_b)
    return __builtin_amdgcn_wmma_f32_16x16x32_bf16(false, a, false, b,
                                                   (short)0, c, false, false);
}

__device__ __forceinline__ float bf2f(u16 u) {
    unsigned v = ((unsigned)u) << 16;
    return __builtin_bit_cast(float, v);
}
__device__ __forceinline__ u16 f2bf(float f) {
    unsigned u = __builtin_bit_cast(unsigned, f);
    u += 0x7FFFu + ((u >> 16) & 1u);          // round-to-nearest-even
    return (u16)(u >> 16);
}

#if __has_builtin(__builtin_amdgcn_cvt_pk_bf16_f32)
typedef __attribute__((ext_vector_type(2))) __bf16 v2bf;
__device__ __forceinline__ unsigned f2bf2(float a, float b) {
    v2bf r = __builtin_amdgcn_cvt_pk_bf16_f32(a, b);   // v_cvt_pk_bf16_f32
    return __builtin_bit_cast(unsigned, r);
}
#else
__device__ __forceinline__ unsigned f2bf2(float a, float b) {
    return (unsigned)f2bf(a) | ((unsigned)f2bf(b) << 16);
}
#endif

#ifdef HAVE_ASYNC
__device__ __forceinline__ void async_b128(const u16* gsrc, u16* ldst) {
    __builtin_amdgcn_global_load_async_to_lds_b128(
        (GLOBAL_AS v4i*)gsrc, (LDS_AS v4i*)ldst, 0, 0);
}
#endif

#ifdef HAVE_TDM
// Issue a 2D TDM tile load: 64 x 64 bf16 tile, row stride `stride_elems`
// in global memory, destination LDS rows padded with 4 dwords per 32 dwords
// (matches a [rows][64+8] u16 LDS array).
__device__ __forceinline__ void tdm_load_tile_64x64(unsigned lds_off,
                                                    unsigned long long gaddr,
                                                    unsigned stride_elems) {
    v4u g0;
    g0[0] = 1u;                                    // count=1, user descriptor
    g0[1] = lds_off;                               // lds_addr (bytes)
    g0[2] = (unsigned)gaddr;                       // global_addr[31:0]
    g0[3] = (unsigned)((gaddr >> 32) & 0x01FFFFFFu) | (2u << 30); // [56:32]|type=2
    v8i g1;
    // data_size=1 (2B) | pad_enable | pad_interval=4 (32 dwords) | pad_amount=3 (4 dwords)
    g1[0] = (int)((1u << 16) | (1u << 20) | (4u << 22) | (3u << 25));
    g1[1] = (int)(64u << 16);                      // tensor_dim0[15:0] = 64
    g1[2] = (int)(64u << 16);                      // td0 hi=0 | tensor_dim1 lo=64
    g1[3] = (int)(64u << 16);                      // td1 hi=0 | tile_dim0=64
    g1[4] = 64;                                    // tile_dim1=64, tile_dim2=0
    g1[5] = (int)stride_elems;                     // tensor_dim0_stride lo32
    g1[6] = 0;
    g1[7] = 0;
    v4i z = {};
#if __clang_major__ >= 23
    v8i z8 = {};
    __builtin_amdgcn_tensor_load_to_lds(g0, g1, z, z, z8, 0);
#else
    __builtin_amdgcn_tensor_load_to_lds(g0, g1, z, z, 0);
#endif
}
#endif

#define PADF (-4294967295.0f)   // float(-2**32 + 1)

// ---------------------------------------------------------------------------
// Kernel 1/3: Out = act(A @ W + bias) * out_scale   (act = relu)
//   A: M x K   (f32 if IN_F32 else bf16-in-u16)
//   W: K x N   f32 (cast to bf16 on the fly)
//   Out: M x N (f32 if OUT_F32 else bf16-in-u16)
// Tiling: 128x128 block tile, K-step 32, 256 threads = 8 waves (2 x 4).
// Each wave: 64(M) x 32(N) -> 4x2 WMMA accumulators.
// ---------------------------------------------------------------------------
template <bool IN_F32, bool OUT_F32>
__global__ __launch_bounds__(256)
void gemm_bias_relu(const void* __restrict__ Ag,
                    const float* __restrict__ W,
                    const float* __restrict__ bias,
                    void* __restrict__ Out,
                    int M, int N, int K, float out_scale)
{
    constexpr int BM = 128, BN = 128, BK = 32, LDA = BK + 8;
    __shared__ u16 As[BM][LDA];   // [m][k]
    __shared__ u16 Bs[BN][LDA];   // [n][k]  (W transposed into LDS)

    const int tid  = threadIdx.x;
    const int wid  = tid >> 5;
    const int lane = tid & 31;
    const int half = lane >> 4;
    const int l16  = lane & 15;
    const int wm   = wid >> 2;          // 0..1
    const int wn   = wid & 3;           // 0..3
    const int m0   = blockIdx.y * BM;
    const int n0   = blockIdx.x * BN;

    v8f acc[4][2];
#pragma unroll
    for (int mi = 0; mi < 4; ++mi)
#pragma unroll
        for (int nf = 0; nf < 2; ++nf)
            acc[mi][nf] = {};

    for (int k0 = 0; k0 < K; k0 += BK) {
        __syncthreads();   // protect LDS reads of previous iteration

        // ---- stage A tile (128 x 32) into LDS as bf16 --------------------
        if (IN_F32) {
            const float* Af = (const float*)Ag;
#pragma unroll
            for (int i = 0; i < 4; ++i) {
                int idx = tid + i * 256;        // 0..1023
                int r   = idx >> 3;             // row 0..127
                int c   = (idx & 7) * 4;        // col 0..28
                float4 vv = *(const float4*)&Af[(size_t)(m0 + r) * K + k0 + c];
                uint2 pk;
                pk.x = f2bf2(vv.x, vv.y);
                pk.y = f2bf2(vv.z, vv.w);
                *(uint2*)&As[r][c] = pk;
            }
            if (k0 + BK < K)
                __builtin_prefetch(&Af[(size_t)(m0 + (tid >> 3)) * K + k0 + BK], 0, 3);
        } else {
            const u16* Ab = (const u16*)Ag;
#ifdef HAVE_ASYNC
#pragma unroll
            for (int i = 0; i < 2; ++i) {
                int idx = tid + i * 256;        // 0..511
                int r   = idx >> 2;             // row 0..127
                int c   = (idx & 3) * 8;        // col 0..24
                async_b128(&Ab[(size_t)(m0 + r) * K + k0 + c], &As[r][c]);
            }
            __builtin_amdgcn_s_wait_asynccnt(0);
#else
#pragma unroll
            for (int i = 0; i < 2; ++i) {
                int idx = tid + i * 256;        // 0..511
                int r   = idx >> 2;             // row 0..127
                int c   = (idx & 3) * 8;        // col 0..24
                *(us8*)&As[r][c] = *(const us8*)&Ab[(size_t)(m0 + r) * K + k0 + c];
            }
#endif
            if (k0 + BK < K)
                __builtin_prefetch(&((const u16*)Ag)[(size_t)(m0 + (tid >> 2)) * K + k0 + BK], 0, 3);
        }

        // ---- stage W tile (32 x 128) transposed into LDS as bf16 ---------
#pragma unroll
        for (int i = 0; i < 4; ++i) {
            int idx = tid + i * 256;            // 0..1023
            int kk  = idx >> 5;                 // k row 0..31
            int nn  = (idx & 31) * 4;           // n col 0..124
            float4 vv = *(const float4*)&W[(size_t)(k0 + kk) * N + n0 + nn];
            Bs[nn + 0][kk] = f2bf(vv.x);
            Bs[nn + 1][kk] = f2bf(vv.y);
            Bs[nn + 2][kk] = f2bf(vv.z);
            Bs[nn + 3][kk] = f2bf(vv.w);
        }
        __syncthreads();

        // ---- fragments + 8 WMMAs ----------------------------------------
        FragBF bfr[2];
#pragma unroll
        for (int nf = 0; nf < 2; ++nf) {
            int n  = wn * 32 + nf * 16 + l16;
            int kb = half * 8;                       // B: K 0-7 / 8-15, then K 16-23 / 24-31
            bfr[nf].h[0] = *(const us8*)&Bs[n][kb];
            bfr[nf].h[1] = *(const us8*)&Bs[n][kb + 16];
        }
        FragBF afr[4];
#pragma unroll
        for (int mi = 0; mi < 4; ++mi) {
            int r = wm * 64 + mi * 16 + l16;         // A: lane holds row l&15
            afr[mi].h[0] = *(const us8*)&As[r][half * 8];        // K 0-7 / 8-15
            afr[mi].h[1] = *(const us8*)&As[r][16 + half * 8];   // K 16-23 / 24-31
        }
#pragma unroll
        for (int mi = 0; mi < 4; ++mi)
#pragma unroll
            for (int nf = 0; nf < 2; ++nf)
                acc[mi][nf] = wmma_bf16(afr[mi].v, bfr[nf].v, acc[mi][nf]);
    }

    // ---- epilogue: bias + relu + scale, store ---------------------------
#pragma unroll
    for (int nf = 0; nf < 2; ++nf) {
        int ng   = n0 + wn * 32 + nf * 16 + l16;
        float bv = bias[ng];
#pragma unroll
        for (int mi = 0; mi < 4; ++mi) {
#pragma unroll
            for (int j = 0; j < 8; ++j) {
                int   mg = m0 + wm * 64 + mi * 16 + half * 8 + j;  // C: M = j + 8*half
                float v  = acc[mi][nf][j] + bv;
                v = fmaxf(v, 0.0f) * out_scale;
                if (OUT_F32)
                    ((float*)Out)[(size_t)mg * N + ng] = v;
                else
                    ((u16*)Out)[(size_t)mg * N + ng] = f2bf(v);
            }
        }
    }
}

// ---------------------------------------------------------------------------
// Kernel 2: flash attention per (head, batch, 128-query tile).
// 8 waves x 16 query rows; key chunks of 64; dk = 64.
// Q already carries the 1/sqrt(D) scale (folded into Q projection).
// K tile staged by the Tensor Data Mover when available.
// ---------------------------------------------------------------------------
__global__ __launch_bounds__(256)
void attention_kernel(const u16* __restrict__ Qp,
                      const u16* __restrict__ Kp,
                      const u16* __restrict__ Vp,
                      u16* __restrict__ Oattn)
{
    constexpr int S = 1024, D = 1024, DK = 64, KC = 64, LDT = DK + 8;
    __shared__ u16 Ksh[KC][LDT];          // [key][d]
    __shared__ u16 Vsh[DK][KC + 8];       // [d][key]  (transposed for B operand)
    __shared__ u16 Psh[8][16][KC + 8];    // per-wave P staging [m][key]
    __shared__ u16 kflag[KC];             // 1 = padded key row

    const int tid  = threadIdx.x;
    const int wid  = tid >> 5;
    const int lane = tid & 31;
    const int half = lane >> 4;
    const int l16  = lane & 15;
    const int head = blockIdx.z;
    const int b    = blockIdx.y;
    const int q0   = blockIdx.x * 128;
    const int qrow = q0 + wid * 16 + l16;

    // ---- load Q fragments (kept in registers for the whole key loop) ----
    const size_t baseQ = ((size_t)b * S + qrow) * D + (size_t)head * DK;
    FragBF qa[2];
#pragma unroll
    for (int f = 0; f < 2; ++f) {
        qa[f].h[0] = *(const us8*)&Qp[baseQ + f * 32 + half * 8];
        qa[f].h[1] = *(const us8*)&Qp[baseQ + f * 32 + 16 + half * 8];
    }
    // query-row mask: sum |q row| over dk (relu'd -> nonneg, sign intact)
    float qsum = 0.0f;
#pragma unroll
    for (int f = 0; f < 2; ++f)
#pragma unroll
        for (int p = 0; p < 2; ++p)
#pragma unroll
            for (int e = 0; e < 8; ++e)
                qsum += fabsf(bf2f(qa[f].h[p][e]));
    qsum += __shfl_xor(qsum, 16);   // combine the two K-halves of the row

    float run_max[8], run_sum[8];
    v8f   Oacc[4];
#pragma unroll
    for (int j = 0; j < 8; ++j) { run_max[j] = -__builtin_inff(); run_sum[j] = 0.0f; }
#pragma unroll
    for (int df = 0; df < 4; ++df) Oacc[df] = {};

    for (int kc = 0; kc < S; kc += KC) {
        __syncthreads();

        // ---- stage V tile transposed (manual; TDM cannot transpose) ------
#pragma unroll
        for (int i = 0; i < 2; ++i) {
            int idx = tid + i * 256;                 // 0..511
            int r   = idx >> 3;                      // key 0..63
            int d8  = (idx & 7) * 8;                 // d 0..56
            size_t g = ((size_t)b * S + kc + r) * D + (size_t)head * DK + d8;
            us8 vv = *(const us8*)&Vp[g];
#pragma unroll
            for (int e = 0; e < 8; ++e) Vsh[d8 + e][r] = vv[e];
        }

        // ---- stage K tile (natural layout) -------------------------------
#ifdef HAVE_TDM
        if (wid == 0) {
            unsigned lds_off = (unsigned)(uintptr_t)&Ksh[0][0];  // low 32 = LDS byte addr
            unsigned long long ga = (unsigned long long)(uintptr_t)
                &Kp[((size_t)b * S + kc) * D + (size_t)head * DK];
            tdm_load_tile_64x64(lds_off, ga, (unsigned)D);
            __builtin_amdgcn_s_wait_tensorcnt(0);
        }
#else
#pragma unroll
        for (int i = 0; i < 2; ++i) {
            int idx = tid + i * 256;
            int r   = idx >> 3;
            int d8  = (idx & 7) * 8;
            size_t g = ((size_t)b * S + kc + r) * D + (size_t)head * DK + d8;
#ifdef HAVE_ASYNC
            async_b128(&Kp[g], &Ksh[r][d8]);
#else
            *(us8*)&Ksh[r][d8] = *(const us8*)&Kp[g];
#endif
        }
#ifdef HAVE_ASYNC
        __builtin_amdgcn_s_wait_asynccnt(0);
#endif
#endif
        __syncthreads();

        // ---- key padding mask -------------------------------------------
        if (tid < KC) {
            float s = 0.0f;
#pragma unroll
            for (int d8 = 0; d8 < DK; d8 += 8) {
                us8 vv = *(const us8*)&Ksh[tid][d8];
#pragma unroll
                for (int e = 0; e < 8; ++e) s += fabsf(bf2f(vv[e]));
            }
            kflag[tid] = (s == 0.0f) ? (u16)1 : (u16)0;
        }
        __syncthreads();

        // ---- scores: (16 x 64) = Q(16x64) . K^T  (8 WMMAs) ---------------
        v8f sc[4];
#pragma unroll
        for (int nf = 0; nf < 4; ++nf) sc[nf] = {};
#pragma unroll
        for (int nf = 0; nf < 4; ++nf) {
            int key = nf * 16 + l16;                 // B operand: lane -> key column
#pragma unroll
            for (int ks = 0; ks < 2; ++ks) {         // dk = 64 -> two K32 steps
                FragBF kb;
                kb.h[0] = *(const us8*)&Ksh[key][ks * 32 + half * 8];
                kb.h[1] = *(const us8*)&Ksh[key][ks * 32 + 16 + half * 8];
                sc[nf] = wmma_bf16(qa[ks].v, kb.v, sc[nf]);
            }
        }

        // ---- mask + online softmax update -------------------------------
        float nmax[8];
#pragma unroll
        for (int j = 0; j < 8; ++j) nmax[j] = run_max[j];
#pragma unroll
        for (int nf = 0; nf < 4; ++nf) {
            bool msk = kflag[nf * 16 + l16] != 0;
#pragma unroll
            for (int j = 0; j < 8; ++j) {
                float s = msk ? PADF : sc[nf][j];
                sc[nf][j] = s;
                nmax[j] = fmaxf(nmax[j], s);
            }
        }
#pragma unroll
        for (int j = 0; j < 8; ++j)
#pragma unroll
            for (int x = 1; x < 16; x <<= 1)
                nmax[j] = fmaxf(nmax[j], __shfl_xor(nmax[j], x));

        float alpha[8], lsum[8];
#pragma unroll
        for (int j = 0; j < 8; ++j) {
            alpha[j]   = __expf(run_max[j] - nmax[j]);
            run_max[j] = nmax[j];
            lsum[j]    = 0.0f;
        }
#pragma unroll
        for (int nf = 0; nf < 4; ++nf)
#pragma unroll
            for (int j = 0; j < 8; ++j) {
                float p = __expf(sc[nf][j] - nmax[j]);
                sc[nf][j] = p;
                lsum[j] += p;
            }
#pragma unroll
        for (int j = 0; j < 8; ++j) {
#pragma unroll
            for (int x = 1; x < 16; x <<= 1)
                lsum[j] += __shfl_xor(lsum[j], x);
            run_sum[j] = run_sum[j] * alpha[j] + lsum[j];
        }
#pragma unroll
        for (int df = 0; df < 4; ++df)
#pragma unroll
            for (int j = 0; j < 8; ++j)
                Oacc[df][j] *= alpha[j];

        // ---- P: C-layout -> A-layout via per-wave LDS slab ---------------
#pragma unroll
        for (int nf = 0; nf < 4; ++nf) {
            int key = nf * 16 + l16;
#pragma unroll
            for (int j = 0; j < 8; ++j)
                Psh[wid][half * 8 + j][key] = f2bf(sc[nf][j]);
        }
        __syncthreads();

        // ---- O += P(16x64) . V(64x64)  (8 WMMAs) -------------------------
#pragma unroll
        for (int ks = 0; ks < 2; ++ks) {
            FragBF pa;
            pa.h[0] = *(const us8*)&Psh[wid][l16][ks * 32 + half * 8];
            pa.h[1] = *(const us8*)&Psh[wid][l16][ks * 32 + 16 + half * 8];
#pragma unroll
            for (int df = 0; df < 4; ++df) {
                int d = df * 16 + l16;
                FragBF vb;
                vb.h[0] = *(const us8*)&Vsh[d][ks * 32 + half * 8];
                vb.h[1] = *(const us8*)&Vsh[d][ks * 32 + 16 + half * 8];
                Oacc[df] = wmma_bf16(pa.v, vb.v, Oacc[df]);
            }
        }
    }

    // ---- finalize: /sum, * qmask, store bf16 ----------------------------
    float qm = (qsum != 0.0f) ? 1.0f : 0.0f;         // valid for row l&15
    float qmC[8], inv[8];
#pragma unroll
    for (int j = 0; j < 8; ++j) {
        qmC[j] = __shfl(qm, half * 8 + j);           // lane m holds row m's mask
        inv[j] = 1.0f / run_sum[j];
    }
#pragma unroll
    for (int df = 0; df < 4; ++df) {
        int dcol = (int)head * DK + df * 16 + l16;
#pragma unroll
        for (int j = 0; j < 8; ++j) {
            int   m = half * 8 + j;                  // C-layout row
            int   q = q0 + wid * 16 + m;
            float v = Oacc[df][j] * inv[j] * qmC[j];
            Oattn[((size_t)b * S + q) * D + dcol] = f2bf(v);
        }
    }
}

// ---------------------------------------------------------------------------
// Launcher
// ---------------------------------------------------------------------------
extern "C" void kernel_launch(void* const* d_in, const int* in_sizes, int n_in,
                              void* d_out, int out_size, void* d_ws, size_t ws_size,
                              hipStream_t stream)
{
    (void)in_sizes; (void)n_in; (void)out_size; (void)ws_size;

    const int B = 8, S = 1024, D = 1024;
    const int M = B * S;                 // 8192

    const float* Q  = (const float*)d_in[0];
    const float* K  = (const float*)d_in[1];
    const float* V  = (const float*)d_in[2];
    const float* WQ = (const float*)d_in[3];
    const float* bQ = (const float*)d_in[4];
    const float* WK = (const float*)d_in[5];
    const float* bK = (const float*)d_in[6];
    const float* WV = (const float*)d_in[7];
    const float* bV = (const float*)d_in[8];
    const float* WO = (const float*)d_in[9];
    const float* bO = (const float*)d_in[10];
    // d_in[11] = h (16), hardcoded

    const size_t elems = (size_t)M * D;  // 8M elements per bf16 tensor
    u16* Qp   = (u16*)d_ws;
    u16* Kp   = Qp + elems;
    u16* Vp   = Kp + elems;
    u16* Attn = Vp + elems;

    dim3 blk(256);
    dim3 gemm_grid(D / 128, M / 128);            // 8 x 64
    dim3 attn_grid(S / 128, B, 16);              // 8 x 8 x 16 (head)

    // projections (fold 1/sqrt(D) = 1/32 into Q)
    gemm_bias_relu<true, false><<<gemm_grid, blk, 0, stream>>>(
        Q, WQ, bQ, Qp, M, D, D, 1.0f / 32.0f);
    gemm_bias_relu<true, false><<<gemm_grid, blk, 0, stream>>>(
        K, WK, bK, Kp, M, D, D, 1.0f);
    gemm_bias_relu<true, false><<<gemm_grid, blk, 0, stream>>>(
        V, WV, bV, Vp, M, D, D, 1.0f);

    attention_kernel<<<attn_grid, blk, 0, stream>>>(Qp, Kp, Vp, Attn);

    // output projection -> f32
    gemm_bias_relu<false, true><<<gemm_grid, blk, 0, stream>>>(
        Attn, WO, bO, (float*)d_out, M, D, D, 1.0f);
}
